// ModuleRelPropX_24790551232617
// MI455X (gfx1250) — matrix operand order repaired and checked by reading, not compiled, not run on hardware
//
#include <hip/hip_runtime.h>

// Problem constants (match reference)
#define BB 2
#define NN 2048
#define KK 256
#define DD 768
#define HH 256
#define LL 64
#define NBUCK 64
#define BK (BB * KK)          // 512
#define REL_PROP 3

typedef __attribute__((ext_vector_type(16))) __bf16 v16bf;
typedef __attribute__((ext_vector_type(8)))  float  v8f;

union BFrag {
    v16bf v;
    uint4 q[2];
    unsigned short s[16];
};

#define WMMA_BF16(a, b, c) \
    __builtin_amdgcn_wmma_f32_16x16x32_bf16(false, (a), false, (b), (short)0, (c), false, false)

__device__ __forceinline__ unsigned short f2bf(float f) {
    union { float f; unsigned u; } v; v.f = f;
    unsigned r = v.u + 0x7FFFu + ((v.u >> 16) & 1u);   // round-to-nearest-even
    return (unsigned short)(r >> 16);
}
__device__ __forceinline__ float sigmoidf(float x) {
    return 1.0f / (1.0f + __expf(-x));
}

// A-fragment (16x32 bf16, M rows = lane&15): per ISA layout lane half selects
// K subgroup; elements are two contiguous 8-elem runs -> two b128 loads.
__device__ __forceinline__ v16bf load_a_frag(const unsigned short* base, int rowStride,
                                             int kbase, int lane) {
    int row = lane & 15, half = lane >> 4;
    const unsigned short* p = base + (size_t)row * rowStride + kbase + 8 * half;
    BFrag f;
    f.q[0] = *reinterpret_cast<const uint4*>(p);
    f.q[1] = *reinterpret_cast<const uint4*>(p + 16);
    return f.v;
}
// B-fragment (32x16 bf16, N col = lane&15) from row-major W[N, Kd]:
// lane's 16 K elements follow the A-frag K order (two 8-elem runs) -> two b128 loads.
__device__ __forceinline__ v16bf load_b_frag(const unsigned short* w, int rowStride,
                                             int ncol, int kbase, int lane) {
    int half = lane >> 4;
    const unsigned short* p = w + (size_t)ncol * rowStride + kbase + 8 * half;
    BFrag f;
    f.q[0] = *reinterpret_cast<const uint4*>(p);
    f.q[1] = *reinterpret_cast<const uint4*>(p + 16);
    return f.v;
}

// ---------------- elementwise convert ----------------
__global__ void cvt_f32_bf16(const float* __restrict__ src,
                             unsigned short* __restrict__ dst, int n) {
    for (int i = blockIdx.x * blockDim.x + threadIdx.x; i < n; i += gridDim.x * blockDim.x)
        dst[i] = f2bf(src[i]);
}

// ---------------- probs = bf16(sigmoid(scores) * mask), 4 elems/thread ----------------
__global__ void probs_kernel(const float* __restrict__ scores,
                             const float* __restrict__ mask,
                             unsigned short* __restrict__ probs, int nquads) {
    int q = blockIdx.x * blockDim.x + threadIdx.x;
    if (q >= nquads) return;
    size_t i4 = (size_t)q * 4;
    float4 s = *reinterpret_cast<const float4*>(scores + i4);
    float mv = mask[i4 / LL];
    unsigned p0 = f2bf(sigmoidf(s.x) * mv);
    unsigned p1 = f2bf(sigmoidf(s.y) * mv);
    unsigned p2 = f2bf(sigmoidf(s.z) * mv);
    unsigned p3 = f2bf(sigmoidf(s.w) * mv);
    uint2 o;
    o.x = p0 | (p1 << 16);
    o.y = p2 | (p3 << 16);
    *reinterpret_cast<uint2*>(probs + i4) = o;
}

// ---------------- projection GEMM: out[m,n] = sum_k A[m,k] W[n,k] + bias[n] ----------------
// A: [M, Kd] bf16 row-major; W: [N, Kd] bf16 row-major. block=128 (4 waves), tile M16 x N64.
__global__ void proj_gemm(const unsigned short* __restrict__ A,
                          const unsigned short* __restrict__ W,
                          const float* __restrict__ bias,
                          float* __restrict__ out, int N, int Kd) {
    int lane = threadIdx.x & 31, wave = threadIdx.x >> 5;
    int nblocks = N / 64;
    int nb = blockIdx.x % nblocks, mb = blockIdx.x / nblocks;
    int nbase = nb * 64 + wave * 16;
    int n = nbase + (lane & 15);
    const unsigned short* Arow = A + (size_t)mb * 16 * Kd;
    v8f acc = {};
    for (int k0 = 0; k0 < Kd; k0 += 32) {
        v16bf a = load_a_frag(Arow, Kd, k0, lane);
        v16bf b = load_b_frag(W, Kd, n, k0, lane);
        acc = WMMA_BF16(a, b, acc);
    }
    float bv = bias[n];
    int half = lane >> 4;
    for (int r = 0; r < 8; ++r) {
        int m = mb * 16 + r + 8 * half;
        out[(size_t)m * N + n] = acc[r] + bv;
    }
}

// ---------------- fused pair-scores ----------------
// One block per (b, i, j-block16). 4 waves each own a 16-wide L subtile.
// h tile (16 j x 32 H) built in LDS per K-step (relu fused), WMMA vs out_w.
__global__ void score_kernel(const float* __restrict__ leftf,
                             const float* __restrict__ rightf,
                             const float* __restrict__ dist_emb,
                             const unsigned short* __restrict__ owb,
                             const float* __restrict__ out_b,
                             const int* __restrict__ span_begin,
                             const int* __restrict__ span_end,
                             float* __restrict__ scores) {
    __shared__ alignas(16) unsigned short hsh[16 * 32];
    __shared__ int bidx[16];
    int lane = threadIdx.x & 31, wave = threadIdx.x >> 5;
    int jb = blockIdx.x & 15;
    int i  = (blockIdx.x >> 4) & (KK - 1);
    int b  = blockIdx.x >> 12;               // / (16 * 256)
    if (threadIdx.x < 16) {
        int j = jb * 16 + threadIdx.x;
        int d = span_begin[b * KK + j] - span_end[b * KK + i];
        d = d < 0 ? -d : d;
        if (d > NBUCK - 1) d = NBUCK - 1;
        bidx[threadIdx.x] = d;
    }
    __syncthreads();
    const float* lrow = leftf + ((size_t)b * KK + i) * HH;
    int nbase = wave * 16;
    int l = nbase + (lane & 15);
    v8f acc = {};
    for (int k0 = 0; k0 < HH; k0 += 32) {
        for (int e = threadIdx.x; e < 16 * 32; e += blockDim.x) {
            int jj = e >> 5, k = e & 31;
            int hk = k0 + k;
            int j = jb * 16 + jj;
            float v = lrow[hk] + rightf[((size_t)b * KK + j) * HH + hk]
                    + dist_emb[bidx[jj] * HH + hk];
            hsh[e] = f2bf(v > 0.f ? v : 0.f);          // fused relu
        }
        __syncthreads();
        v16bf a  = load_a_frag(hsh, 32, 0, lane);
        v16bf bf = load_b_frag(owb, HH, l, k0, lane);
        acc = WMMA_BF16(a, bf, acc);
        __syncthreads();
    }
    float bv = out_b[l];
    int half = lane >> 4;
    for (int r = 0; r < 8; ++r) {
        int j = jb * 16 + r + 8 * half;
        scores[(((size_t)b * KK + i) * KK + j) * LL + l] = acc[r] + bv;
    }
}

// ---------------- ctxt1: sum_i (probs[b,i,jblk,:] @ A_wT)[j,d] * update[b,i,d] ----------------
// Block = (b, j-block16, d-block64). A frags come straight from the bf16 probs tensor
// (row stride L); B frags (A_w columns) are loop-invariant. No LDS, no barriers.
__global__ void ctxt1_kernel(const unsigned short* __restrict__ probs,
                             const unsigned short* __restrict__ awb,
                             const float* __restrict__ update,
                             const float* __restrict__ span_lengths,
                             float* __restrict__ cbuf) {
    int lane = threadIdx.x & 31, wave = threadIdx.x >> 5;
    int db = blockIdx.x % (DD / 64);
    int jb = (blockIdx.x / (DD / 64)) & 15;
    int b  = blockIdx.x / ((DD / 64) * 16);
    int dglob = db * 64 + wave * 16 + (lane & 15);
    v16bf b0 = load_b_frag(awb, LL, dglob, 0, lane);   // invariant over i
    v16bf b1 = load_b_frag(awb, LL, dglob, 32, lane);
    float accr[8];
    for (int r = 0; r < 8; ++r) accr[r] = 0.f;
    const unsigned short* pb = probs + ((size_t)b * KK * KK + jb * 16) * LL;
    const float* ub = update + (size_t)b * KK * DD + dglob;
    for (int i = 0; i < KK; ++i) {
        const unsigned short* pa = pb + (size_t)i * KK * LL;   // rows jj, stride LL
        v8f w = {};
        v16bf a0 = load_a_frag(pa, LL, 0, lane);
        w = WMMA_BF16(a0, b0, w);
        v16bf a1 = load_a_frag(pa, LL, 32, lane);
        w = WMMA_BF16(a1, b1, w);
        float uv = ub[(size_t)i * DD];
        for (int r = 0; r < 8; ++r) accr[r] += w[r] * uv;
        if (i + 1 < KK) __builtin_prefetch(pa + (size_t)KK * LL, 0, 1);
    }
    float inv = 1.0f / span_lengths[b];
    int half = lane >> 4;
    for (int r = 0; r < 8; ++r) {
        int j = jb * 16 + r + 8 * half;
        cbuf[((size_t)b * KK + j) * DD + dglob] = accr[r] * inv;
    }
}

// ---------------- ctxt2: sum_j (probs[b,iblk,j,:] @ B_wT)[i,d] * update[b,j,d]; adds into cbuf ----
__global__ void ctxt2_kernel(const unsigned short* __restrict__ probs,
                             const unsigned short* __restrict__ bwb,
                             const float* __restrict__ update,
                             const float* __restrict__ span_lengths,
                             float* __restrict__ cbuf) {
    int lane = threadIdx.x & 31, wave = threadIdx.x >> 5;
    int db = blockIdx.x % (DD / 64);
    int ib = (blockIdx.x / (DD / 64)) & 15;
    int b  = blockIdx.x / ((DD / 64) * 16);
    int dglob = db * 64 + wave * 16 + (lane & 15);
    v16bf b0 = load_b_frag(bwb, LL, dglob, 0, lane);   // invariant over j
    v16bf b1 = load_b_frag(bwb, LL, dglob, 32, lane);
    float accr[8];
    for (int r = 0; r < 8; ++r) accr[r] = 0.f;
    const unsigned short* pb = probs + ((size_t)b * KK + ib * 16) * KK * LL;
    const float* ub = update + (size_t)b * KK * DD + dglob;
    for (int j = 0; j < KK; ++j) {
        const unsigned short* pa = pb + (size_t)j * LL;        // rows ii, stride KK*LL
        v8f w = {};
        v16bf a0 = load_a_frag(pa, KK * LL, 0, lane);
        w = WMMA_BF16(a0, b0, w);
        v16bf a1 = load_a_frag(pa, KK * LL, 32, lane);
        w = WMMA_BF16(a1, b1, w);
        float uv = ub[(size_t)j * DD];
        for (int r = 0; r < 8; ++r) accr[r] += w[r] * uv;
        if (j + 1 < KK) __builtin_prefetch(pa + LL, 0, 1);
    }
    float inv = 1.0f / span_lengths[b];
    int half = lane >> 4;
    for (int r = 0; r < 8; ++r) {
        int i = ib * 16 + r + 8 * half;
        cbuf[((size_t)b * KK + i) * DD + dglob] += accr[r] * inv;
    }
}

// ---------------- gate: g = sig([u,c] @ gate_wT + b); u' = g*u + (1-g)*c ----------------
// K-loop over 1536 split between the two bf16 sources (768 each, step-aligned).
__global__ void gate_kernel(const unsigned short* __restrict__ ubf,
                            const unsigned short* __restrict__ cbf,
                            const unsigned short* __restrict__ gwb,
                            const float* __restrict__ gate_b,
                            float* __restrict__ update,
                            const float* __restrict__ cf,
                            unsigned short* __restrict__ ubf_next) {
    int lane = threadIdx.x & 31, wave = threadIdx.x >> 5;
    int db = blockIdx.x % (DD / 64);
    int mb = blockIdx.x / (DD / 64);
    int n = db * 64 + wave * 16 + (lane & 15);
    v8f acc = {};
    for (int k0 = 0; k0 < 2 * DD; k0 += 32) {
        const unsigned short* Ab = (k0 < DD) ? ubf : cbf;
        int kofs = (k0 < DD) ? k0 : (k0 - DD);
        v16bf a  = load_a_frag(Ab + (size_t)mb * 16 * DD, DD, kofs, lane);
        v16bf bf = load_b_frag(gwb, 2 * DD, n, k0, lane);
        acc = WMMA_BF16(a, bf, acc);
    }
    float gb = gate_b[n];
    int half = lane >> 4;
    for (int r = 0; r < 8; ++r) {
        int m = mb * 16 + r + 8 * half;
        size_t idx = (size_t)m * DD + n;
        float g = sigmoidf(acc[r] + gb);
        float u = update[idx], c = cf[idx];
        float nu = g * u + (1.f - g) * c;
        update[idx]   = nu;
        ubf_next[idx] = f2bf(nu);
    }
}

// ---------------- overwrite_spans scatter ----------------
__global__ void scatter_kernel(const float* __restrict__ update,
                               const int* __restrict__ prune,
                               const float* __restrict__ span_lengths,
                               float* __restrict__ out_all) {
    int idx = blockIdx.x * blockDim.x + threadIdx.x;
    if (idx >= BK * DD) return;
    int d  = idx % DD;
    int bk = idx / DD;
    int b = bk / KK, k = bk % KK;
    if ((float)k < span_lengths[b]) {
        int row = prune[bk];
        out_all[((size_t)b * NN + row) * DD + d] = update[idx];
    }
}

extern "C" void kernel_launch(void* const* d_in, const int* in_sizes, int n_in,
                              void* d_out, int out_size, void* d_ws, size_t ws_size,
                              hipStream_t stream) {
    (void)in_sizes; (void)n_in; (void)out_size; (void)ws_size;
    const float* all_span     = (const float*)d_in[0];
    const float* span_vecs    = (const float*)d_in[1];
    const int*   span_begin   = (const int*)d_in[2];
    const int*   span_end     = (const int*)d_in[3];
    const float* square_mask  = (const float*)d_in[4];
    const float* span_lengths = (const float*)d_in[5];
    const int*   prune        = (const int*)d_in[6];
    const float* left_w  = (const float*)d_in[7];
    const float* left_b  = (const float*)d_in[8];
    const float* right_w = (const float*)d_in[9];
    const float* right_b = (const float*)d_in[10];
    const float* dist_emb = (const float*)d_in[11];
    const float* out_w   = (const float*)d_in[12];
    const float* out_b   = (const float*)d_in[13];
    const float* A_w     = (const float*)d_in[14];
    const float* B_w     = (const float*)d_in[15];
    const float* gate_w  = (const float*)d_in[16];
    const float* gate_b  = (const float*)d_in[17];

    float* out_all = (float*)d_out;                       // [B,N,D]
    float* update  = out_all + (size_t)BB * NN * DD;      // [B,K,D]
    float* scores  = update + (size_t)BK * DD;            // [B,K,K,L]

    // workspace sub-allocator (256B aligned)
    size_t off = 0;
    auto alloc = [&](size_t bytes) -> void* {
        off = (off + 255) & ~(size_t)255;
        void* p = (char*)d_ws + off;
        off += bytes;
        return p;
    };
    unsigned short* probsb = (unsigned short*)alloc((size_t)BB * KK * KK * LL * 2);
    unsigned short* ubf0 = (unsigned short*)alloc((size_t)BK * DD * 2);
    unsigned short* ubf1 = (unsigned short*)alloc((size_t)BK * DD * 2);
    unsigned short* cbf  = (unsigned short*)alloc((size_t)BK * DD * 2);
    unsigned short* lwb  = (unsigned short*)alloc((size_t)HH * DD * 2);
    unsigned short* rwb  = (unsigned short*)alloc((size_t)HH * DD * 2);
    unsigned short* owb  = (unsigned short*)alloc((size_t)LL * HH * 2);
    unsigned short* awb  = (unsigned short*)alloc((size_t)DD * LL * 2);
    unsigned short* bwb  = (unsigned short*)alloc((size_t)DD * LL * 2);
    unsigned short* gwb  = (unsigned short*)alloc((size_t)DD * 2 * DD * 2);
    float* leftf  = (float*)alloc((size_t)BK * HH * 4);
    float* rightf = (float*)alloc((size_t)BK * HH * 4);
    float* cf32   = (float*)alloc((size_t)BK * DD * 4);

    auto cvt = [&](const float* s, unsigned short* d, int n) {
        cvt_f32_bf16<<<(n + 255) / 256, 256, 0, stream>>>(s, d, n);
    };

    // weights -> bf16 (once per launch; cheap)
    cvt(left_w,  lwb, HH * DD);
    cvt(right_w, rwb, HH * DD);
    cvt(out_w,   owb, LL * HH);
    cvt(A_w,     awb, DD * LL);
    cvt(B_w,     bwb, DD * LL);
    cvt(gate_w,  gwb, DD * 2 * DD);

    // init outputs / live state
    hipMemcpyAsync(out_all, all_span, sizeof(float) * (size_t)BB * NN * DD,
                   hipMemcpyDeviceToDevice, stream);
    hipMemcpyAsync(update, span_vecs, sizeof(float) * (size_t)BK * DD,
                   hipMemcpyDeviceToDevice, stream);
    cvt(span_vecs, ubf0, BK * DD);

    unsigned short* ucur = ubf0;
    unsigned short* unext = ubf1;

    const int projGrid  = (BK / 16) * (HH / 64);           // 128
    const int scoreGrid = BB * KK * (KK / 16);             // 8192
    const int ctxtGrid  = BB * 16 * (DD / 64);             // 384
    const int gateGrid  = (BK / 16) * (DD / 64);           // 384
    const int nquads    = BB * KK * KK * LL / 4;           // 2M

    auto run_scores = [&]() {
        proj_gemm<<<projGrid, 128, 0, stream>>>(ucur, lwb, left_b,  leftf,  HH, DD);
        proj_gemm<<<projGrid, 128, 0, stream>>>(ucur, rwb, right_b, rightf, HH, DD);
        score_kernel<<<scoreGrid, 128, 0, stream>>>(leftf, rightf, dist_emb, owb, out_b,
                                                    span_begin, span_end, scores);
    };

    run_scores();
    for (int it = 0; it < REL_PROP; ++it) {
        probs_kernel<<<(nquads + 255) / 256, 256, 0, stream>>>(scores, square_mask,
                                                               probsb, nquads);
        ctxt1_kernel<<<ctxtGrid, 128, 0, stream>>>(probsb, awb, update, span_lengths, cf32);
        ctxt2_kernel<<<ctxtGrid, 128, 0, stream>>>(probsb, bwb, update, span_lengths, cf32);
        cvt(cf32, cbf, BK * DD);
        gate_kernel<<<gateGrid, 128, 0, stream>>>(ucur, cbf, gwb, gate_b, update, cf32, unext);
        unsigned short* t = ucur; ucur = unext; unext = t;
        run_scores();
    }

    scatter_kernel<<<(BK * DD + 255) / 256, 256, 0, stream>>>(update, prune,
                                                              span_lengths, out_all);
}
